// GATModel_37641093382932
// MI455X (gfx1250) — matrix operand order repaired and checked by reading, not compiled, not run on hardware
//
#include <hip/hip_runtime.h>

typedef __attribute__((ext_vector_type(16))) _Float16 v16h;
typedef __attribute__((ext_vector_type(8)))  _Float16 v8h;
typedef __attribute__((ext_vector_type(8)))  float    v8f;

static constexpr int N_NODES = 100000;
static constexpr int N_EDGES = 1600000;
static constexpr int N_TOT_E = N_EDGES + N_NODES;   // edges + self loops
static constexpr int NBATCH  = 512;
static constexpr int DIN0    = 256;
static constexpr int DH1     = 128;
static constexpr int DH2     = 64;
static constexpr int NCLS    = 27;
static constexpr float SLOPE = 0.2f;

// ---------------- atomic helpers ----------------
__device__ __forceinline__ void atomicAddF(float* p, float v) {
  unsafeAtomicAdd(p, v);   // native global_atomic_add_f32 (no CAS loop)
}

__device__ __forceinline__ void atomicMaxF(float* p, float v) {
  // signed-int max for v>=0, unsigned-int min for v<0; init must be -inf
  if (v >= 0.0f) atomicMax((int*)p, __float_as_int(v));
  else           atomicMin((unsigned int*)p, __float_as_uint(v));
}

// ---------------- elementwise ----------------
__global__ void k_fill(float* __restrict__ p, float v, int n) {
  int i = blockIdx.x * blockDim.x + threadIdx.x;
  if (i < n) p[i] = v;
}

__global__ void k_fill_bias(float* __restrict__ out, const float* __restrict__ b,
                            int n, int dout) {
  int i = blockIdx.x * blockDim.x + threadIdx.x;
  if (i < n) out[i] = b[i & (dout - 1)];   // dout is a power of two
}

__global__ void k_f32_to_f16(const float* __restrict__ in, _Float16* __restrict__ out, int n) {
  int i = blockIdx.x * blockDim.x + threadIdx.x;
  if (i < n) out[i] = (_Float16)in[i];
}

// Wt[n][k] = (f16) W[k][n] ; W is [din, dout] row-major
__global__ void k_transpose_w(const float* __restrict__ w, _Float16* __restrict__ wt,
                              int din, int dout) {
  int i = blockIdx.x * blockDim.x + threadIdx.x;
  if (i >= din * dout) return;
  int k = i / dout, n = i % dout;
  wt[(size_t)n * din + k] = (_Float16)w[i];
}

// ---------------- WMMA GEMM:  h[N,DOUT] = xh[N,DIN](f16) * Wt[DOUT,DIN](f16)^T ----------------
// One wave computes one 16x16 output tile. 8 waves / block.
template<int DIN, int DOUT>
__global__ void wmma_gemm(const _Float16* __restrict__ xh,
                          const _Float16* __restrict__ wt,
                          float* __restrict__ h, int nrows) {
  static_assert(DIN % 32 == 0, "");
  constexpr int COL_TILES = DOUT / 16;              // 8 (DH1) or 4 (DH2)
  constexpr int ROW_TILES_PER_BLOCK = 8 / COL_TILES;
  const int wave = threadIdx.x >> 5;
  const int lane = threadIdx.x & 31;
  const int col_tile = wave % COL_TILES;
  const int row_tile = blockIdx.x * ROW_TILES_PER_BLOCK + wave / COL_TILES;
  const int row_base = row_tile * 16;
  if (row_base >= nrows) return;

  const int half = lane >> 4;      // K-half select
  const int m    = lane & 15;      // A-row within tile / B-column

  const _Float16* __restrict__ arow = xh + (size_t)(row_base + m) * DIN;
  const _Float16* __restrict__ brow = wt + (size_t)(col_tile * 16 + m) * DIN;

  v8f acc = {};
  for (int k = 0; k < DIN; k += 32) {
    // A 16x32 f16 fragment: lane holds K = {8h..8h+7, 16+8h..16+8h+7} of row m
    union { v16h v; v8h h2[2]; } au;
    au.h2[0] = *(const v8h*)(arow + k + 8 * half);
    au.h2[1] = *(const v8h*)(arow + k + 16 + 8 * half);
    // B 32x16 f16 fragment: lane holds K = {8h..8h+7, 16+8h..16+8h+7} of column m (W stored transposed)
    union { v16h v; v8h h2[2]; } bu;
    bu.h2[0] = *(const v8h*)(brow + k + 8 * half);
    bu.h2[1] = *(const v8h*)(brow + k + 16 + 8 * half);
    v16h bv = bu.v;
    acc = __builtin_amdgcn_wmma_f32_16x16x32_f16(false, au.v, false, bv,
                                                 (short)0, acc, false, false);
  }
  // D: lane holds rows r + 8*half, col m
#pragma unroll
  for (int r = 0; r < 8; ++r)
    h[(size_t)(row_base + r + 8 * half) * DOUT + col_tile * 16 + m] = acc[r];
}

// ---------------- attention logits per node: al = h . a ----------------
__global__ void k_compute_al(const float* __restrict__ h,
                             const float* __restrict__ a_src, const float* __restrict__ a_dst,
                             float* __restrict__ al_s, float* __restrict__ al_d,
                             int n, int dout) {
  int row  = blockIdx.x * (blockDim.x >> 5) + (threadIdx.x >> 5);
  int lane = threadIdx.x & 31;
  if (row >= n) return;
  float ss = 0.f, sd = 0.f;
  for (int c = lane; c < dout; c += 32) {
    float v = h[(size_t)row * dout + c];
    ss += v * a_src[c];
    sd += v * a_dst[c];
  }
  for (int o = 16; o; o >>= 1) {
    ss += __shfl_down(ss, o, 32);
    sd += __shfl_down(sd, o, 32);
  }
  if (lane == 0) { al_s[row] = ss; al_d[row] = sd; }
}

// ---------------- edge pass 1: leaky-relu logit + segment max ----------------
__global__ void k_edge_logit_max(const int* __restrict__ srcI, const int* __restrict__ dstI,
                                 const float* __restrict__ al_s, const float* __restrict__ al_d,
                                 float* __restrict__ elogit, float* __restrict__ mseg,
                                 int E, int total) {
  int i = blockIdx.x * blockDim.x + threadIdx.x;
  if (i >= total) return;
  int s = (i < E) ? srcI[i] : (i - E);
  int d = (i < E) ? dstI[i] : (i - E);
  float e = al_s[s] + al_d[d];
  e = (e >= 0.f) ? e : SLOPE * e;
  elogit[i] = e;
  atomicMaxF(&mseg[d], e);
}

// ---------------- edge pass 2: exp + segment sum ----------------
__global__ void k_edge_expsum(const int* __restrict__ dstI,
                              const float* __restrict__ elogit, const float* __restrict__ mseg,
                              float* __restrict__ ew, float* __restrict__ denom,
                              int E, int total) {
  int i = blockIdx.x * blockDim.x + threadIdx.x;
  if (i >= total) return;
  int d = (i < E) ? dstI[i] : (i - E);
  float w = __expf(elogit[i] - mseg[d]);
  ew[i] = w;
  atomicAddF(&denom[d], w);
}

// ---------------- edge pass 3: weighted scatter-add (one wave per edge) ----------------
__global__ void k_edge_aggregate(const int* __restrict__ srcI, const int* __restrict__ dstI,
                                 const float* __restrict__ ew, const float* __restrict__ denom,
                                 const float* __restrict__ h, float* __restrict__ out,
                                 int E, int total, int dout) {
  int wid  = blockIdx.x * (blockDim.x >> 5) + (threadIdx.x >> 5);
  int lane = threadIdx.x & 31;
  if (wid >= total) return;
  int s = (wid < E) ? srcI[wid] : (wid - E);
  int d = (wid < E) ? dstI[wid] : (wid - E);
  float alpha = ew[wid] / denom[d];
  const float* __restrict__ hs = h + (size_t)s * dout;
  float* __restrict__ od = out + (size_t)d * dout;
  for (int c = lane; c < dout; c += 32)
    atomicAddF(&od[c], hs[c] * alpha);
}

// ---------------- mean pool (accumulate) ----------------
__global__ void k_pool_acc(const float* __restrict__ x, const int* __restrict__ batch,
                           float* __restrict__ pooled, float* __restrict__ cnt,
                           int n, int dout) {
  int wid  = blockIdx.x * (blockDim.x >> 5) + (threadIdx.x >> 5);
  int lane = threadIdx.x & 31;
  if (wid >= n) return;
  int g = batch[wid];
  for (int c = lane; c < dout; c += 32)
    atomicAddF(&pooled[(size_t)g * dout + c], x[(size_t)wid * dout + c]);
  if (lane == 0) atomicAddF(&cnt[g], 1.0f);
}

// ---------------- final linear + sigmoid ----------------
__global__ void k_final(const float* __restrict__ ps, const float* __restrict__ pt,
                        const float* __restrict__ cs, const float* __restrict__ ct,
                        const float* __restrict__ lw, const float* __restrict__ lb,
                        float* __restrict__ out) {
  int g = blockIdx.x;
  int lane = threadIdx.x;
  __shared__ float sv[DH2];
  float invs = 1.0f / fmaxf(cs[g], 1.0f);
  float invt = 1.0f / fmaxf(ct[g], 1.0f);
  for (int k = lane; k < DH2; k += 32)
    sv[k] = ps[(size_t)g * DH2 + k] * invs + pt[(size_t)g * DH2 + k] * invt;
  __syncthreads();
  for (int c = lane; c < NCLS; c += 32) {
    float acc = lb[c];
    for (int k = 0; k < DH2; ++k) acc += sv[k] * lw[k * NCLS + c];
    out[g * NCLS + c] = 1.0f / (1.0f + __expf(-acc));
  }
}

// ---------------- host-side branch driver ----------------
static void run_branch(const float* x, const int* ei, const int* batch,
                       const float* W1, const float* as1, const float* ad1, const float* b1,
                       const float* W2, const float* as2, const float* ad2, const float* b2,
                       float* pooled, float* cnt,
                       char* bufA, float* bufB, float* bufC,
                       float* al_s, float* al_d, float* mseg, float* denom,
                       float* elogit, float* ew, _Float16* wtbuf,
                       hipStream_t stream) {
  const int TB = 256;
  const int gElemFeat1 = (N_NODES * DIN0 + TB - 1) / TB;
  const int gElemH1    = (N_NODES * DH1 + TB - 1) / TB;
  const int gElemH2    = (N_NODES * DH2 + TB - 1) / TB;
  const int gNode      = (N_NODES + TB - 1) / TB;
  const int gEdge      = (N_TOT_E + TB - 1) / TB;
  const int gWaveEdge  = (N_TOT_E + 7) / 8;     // 8 waves / block
  const int gWaveNode  = (N_NODES + 7) / 8;

  // ===== layer 1 : din=256, dout=128 =====
  _Float16* xh = (_Float16*)bufA;                      // [N,256] f16
  float*    h1 = bufB;                                 // [N,128] f32
  float*    x1 = bufC;                                 // [N,128] f32 (agg out)

  k_f32_to_f16<<<gElemFeat1, TB, 0, stream>>>(x, xh, N_NODES * DIN0);
  k_transpose_w<<<(DIN0 * DH1 + TB - 1) / TB, TB, 0, stream>>>(W1, wtbuf, DIN0, DH1);
  wmma_gemm<DIN0, DH1><<<N_NODES / 16, TB, 0, stream>>>(xh, wtbuf, h1, N_NODES);
  k_compute_al<<<gWaveNode, TB, 0, stream>>>(h1, as1, ad1, al_s, al_d, N_NODES, DH1);
  k_fill<<<gNode, TB, 0, stream>>>(mseg, -INFINITY, N_NODES);
  k_fill<<<gNode, TB, 0, stream>>>(denom, 0.0f, N_NODES);
  k_fill_bias<<<gElemH1, TB, 0, stream>>>(x1, b1, N_NODES * DH1, DH1);
  k_edge_logit_max<<<gEdge, TB, 0, stream>>>(ei, ei + N_EDGES, al_s, al_d, elogit, mseg,
                                             N_EDGES, N_TOT_E);
  k_edge_expsum<<<gEdge, TB, 0, stream>>>(ei + N_EDGES, elogit, mseg, ew, denom,
                                          N_EDGES, N_TOT_E);
  k_edge_aggregate<<<gWaveEdge, TB, 0, stream>>>(ei, ei + N_EDGES, ew, denom, h1, x1,
                                                 N_EDGES, N_TOT_E, DH1);

  // ===== layer 2 : din=128, dout=64 (reuse bufA for x1h + h2, bufB for x2) =====
  _Float16* x1h = (_Float16*)bufA;                                  // [N,128] f16
  float*    h2  = (float*)(bufA + (size_t)N_NODES * DH1 * sizeof(_Float16)); // [N,64] f32
  float*    x2  = bufB;                                             // [N,64] f32

  k_f32_to_f16<<<gElemH1, TB, 0, stream>>>(x1, x1h, N_NODES * DH1);
  k_transpose_w<<<(DH1 * DH2 + TB - 1) / TB, TB, 0, stream>>>(W2, wtbuf, DH1, DH2);
  wmma_gemm<DH1, DH2><<<N_NODES / 32, TB, 0, stream>>>(x1h, wtbuf, h2, N_NODES);
  k_compute_al<<<gWaveNode, TB, 0, stream>>>(h2, as2, ad2, al_s, al_d, N_NODES, DH2);
  k_fill<<<gNode, TB, 0, stream>>>(mseg, -INFINITY, N_NODES);
  k_fill<<<gNode, TB, 0, stream>>>(denom, 0.0f, N_NODES);
  k_fill_bias<<<gElemH2, TB, 0, stream>>>(x2, b2, N_NODES * DH2, DH2);
  k_edge_logit_max<<<gEdge, TB, 0, stream>>>(ei, ei + N_EDGES, al_s, al_d, elogit, mseg,
                                             N_EDGES, N_TOT_E);
  k_edge_expsum<<<gEdge, TB, 0, stream>>>(ei + N_EDGES, elogit, mseg, ew, denom,
                                          N_EDGES, N_TOT_E);
  k_edge_aggregate<<<gWaveEdge, TB, 0, stream>>>(ei, ei + N_EDGES, ew, denom, h2, x2,
                                                 N_EDGES, N_TOT_E, DH2);

  // ===== mean pool =====
  k_fill<<<(NBATCH * DH2 + TB - 1) / TB, TB, 0, stream>>>(pooled, 0.0f, NBATCH * DH2);
  k_fill<<<(NBATCH + TB - 1) / TB, TB, 0, stream>>>(cnt, 0.0f, NBATCH);
  k_pool_acc<<<gWaveNode, TB, 0, stream>>>(x2, batch, pooled, cnt, N_NODES, DH2);
}

extern "C" void kernel_launch(void* const* d_in, const int* in_sizes, int n_in,
                              void* d_out, int out_size, void* d_ws, size_t ws_size,
                              hipStream_t stream) {
  // inputs in setup_inputs() dict order
  const float* x_s  = (const float*)d_in[0];
  const float* x_t  = (const float*)d_in[1];
  const int*   ei_s = (const int*)d_in[2];
  const int*   ei_t = (const int*)d_in[3];
  const int*   b_s  = (const int*)d_in[4];
  const int*   b_t  = (const int*)d_in[5];
  const float *W_s1 = (const float*)d_in[6],  *as_s1 = (const float*)d_in[7],
              *ad_s1 = (const float*)d_in[8], *bb_s1 = (const float*)d_in[9];
  const float *W_s2 = (const float*)d_in[10], *as_s2 = (const float*)d_in[11],
              *ad_s2 = (const float*)d_in[12], *bb_s2 = (const float*)d_in[13];
  const float *W_t1 = (const float*)d_in[14], *as_t1 = (const float*)d_in[15],
              *ad_t1 = (const float*)d_in[16], *bb_t1 = (const float*)d_in[17];
  const float *W_t2 = (const float*)d_in[18], *as_t2 = (const float*)d_in[19],
              *ad_t2 = (const float*)d_in[20], *bb_t2 = (const float*)d_in[21];
  const float* lin_w = (const float*)d_in[22];
  const float* lin_b = (const float*)d_in[23];
  float* out = (float*)d_out;

  // ---- workspace bump allocator (256B aligned) ----
  char* ws = (char*)d_ws;
  size_t off = 0;
  auto alloc = [&](size_t bytes) -> void* {
    void* p = ws + off;
    off += (bytes + 255) & ~(size_t)255;
    return p;
  };
  char*  bufA   = (char*) alloc((size_t)N_NODES * DIN0 * sizeof(_Float16)); // 51.2MB
  float* bufB   = (float*)alloc((size_t)N_NODES * DH1 * sizeof(float));     // 51.2MB
  float* bufC   = (float*)alloc((size_t)N_NODES * DH1 * sizeof(float));     // 51.2MB
  float* al_s   = (float*)alloc((size_t)N_NODES * sizeof(float));
  float* al_d   = (float*)alloc((size_t)N_NODES * sizeof(float));
  float* mseg   = (float*)alloc((size_t)N_NODES * sizeof(float));
  float* denom  = (float*)alloc((size_t)N_NODES * sizeof(float));
  float* elogit = (float*)alloc((size_t)N_TOT_E * sizeof(float));
  float* ew     = (float*)alloc((size_t)N_TOT_E * sizeof(float));
  _Float16* wtbuf = (_Float16*)alloc((size_t)DIN0 * DH1 * sizeof(_Float16));
  float* pooled_s = (float*)alloc((size_t)NBATCH * DH2 * sizeof(float));
  float* pooled_t = (float*)alloc((size_t)NBATCH * DH2 * sizeof(float));
  float* cnt_s    = (float*)alloc((size_t)NBATCH * sizeof(float));
  float* cnt_t    = (float*)alloc((size_t)NBATCH * sizeof(float));
  (void)ws_size; (void)n_in; (void)in_sizes; (void)out_size;

  run_branch(x_s, ei_s, b_s, W_s1, as_s1, ad_s1, bb_s1, W_s2, as_s2, ad_s2, bb_s2,
             pooled_s, cnt_s, bufA, bufB, bufC, al_s, al_d, mseg, denom,
             elogit, ew, wtbuf, stream);
  run_branch(x_t, ei_t, b_t, W_t1, as_t1, ad_t1, bb_t1, W_t2, as_t2, ad_t2, bb_t2,
             pooled_t, cnt_t, bufA, bufB, bufC, al_s, al_d, mseg, denom,
             elogit, ew, wtbuf, stream);

  k_final<<<NBATCH, 32, 0, stream>>>(pooled_s, pooled_t, cnt_s, cnt_t, lin_w, lin_b, out);
}
